// LSTM_RNN_60232621359354
// MI455X (gfx1250) — matrix operand order repaired and checked by reading, not compiled, not run on hardware
//
#include <hip/hip_runtime.h>

#define T_STEPS 256
#define BATCH   128
#define DIM     512
#define HID     512
#define FH      2048   // 4*HID

typedef __attribute__((ext_vector_type(16))) _Float16 v16h;
typedef __attribute__((ext_vector_type(8)))  float    v8f;
typedef __attribute__((ext_vector_type(4)))  int      i4;   // trivial 16-byte chunk

__device__ __forceinline__ float sigmoidf_(float x) {
    return 1.0f / (1.0f + __expf(-x));
}

// A fragment (16x32 f16, M=batch rows): lanes 0-15 -> M=lane, K = kc+0..7 & kc+16..23
//                                       lanes 16-31 -> M=lane-16, K = kc+8..15 & kc+24..31
__device__ __forceinline__ v16h load_a_frag(const _Float16* base, int kc, int lane) {
    const int m    = lane & 15;
    const int koff = (lane >> 4) * 8;
    const _Float16* p = base + m * DIM + kc + koff;
    union { v16h v; i4 q[2]; } u;
    u.q[0] = *(const i4*)(p);        // K = kc+koff   .. +7
    u.q[1] = *(const i4*)(p + 16);   // K = kc+16+koff.. +7
    return u.v;
}

// B fragment (32x16 f16): column n of B == row (nBase+n) of W (row-major [FH][DIM]).
// lanes 0-15 hold K=kc+0..7 & kc+16..23 of column n; lanes 16-31 hold K=kc+8..15 & kc+24..31.
__device__ __forceinline__ v16h load_b_frag(const _Float16* w, int nBase, int kc, int lane) {
    const int n    = lane & 15;
    const int koff = (lane >> 4) * 8;
    const _Float16* p = w + (size_t)(nBase + n) * DIM + kc + koff;
    union { v16h v; i4 q[2]; } u;
    u.q[0] = *(const i4*)(p);        // K = kc+koff   .. +7
    u.q[1] = *(const i4*)(p + 16);   // K = kc+16+koff.. +7
    return u.v;
}

// One-time prep: f32 weights -> f16 (B-operand friendly, row-major), fused bias.
__global__ __launch_bounds__(256) void lstm_prep_kernel(
    const float* __restrict__ wih, const float* __restrict__ whh,
    const float* __restrict__ bih, const float* __restrict__ bhh,
    _Float16* __restrict__ wih16, _Float16* __restrict__ whh16,
    float* __restrict__ bias)
{
    const int stride = gridDim.x * blockDim.x;
    const int i0 = blockIdx.x * blockDim.x + threadIdx.x;
    const int NW = FH * DIM;
    for (int i = i0; i < NW; i += stride) {
        wih16[i] = (_Float16)wih[i];
        whh16[i] = (_Float16)whh[i];
    }
    for (int i = i0; i < FH; i += stride)
        bias[i] = bih[i] + bhh[i];
}

__global__ __launch_bounds__(512, 1) void lstm_fused_kernel(
    const float* __restrict__ feat,       // [T,B,D]
    const float* __restrict__ hx0,        // [B,H]
    const float* __restrict__ cx0,        // [B,H]
    const int*   __restrict__ mask,       // [T,B,1]
    const _Float16* __restrict__ wih_in,  // [4H,D] f16
    const _Float16* __restrict__ whh_in,  // [4H,H] f16
    const float* __restrict__ bias,       // [4H]
    float* __restrict__ hs_out,           // [T,B,H]
    float* __restrict__ cs_out)           // [T,B,H]
{
    __shared__ _Float16 xls[16 * DIM];    // x tile, f16
    __shared__ _Float16 hls[16 * DIM];    // h tile, f16 (recurrent state)
    __shared__ float    keeps[16];

    const int tid  = threadIdx.x;
    const int lane = tid & 31;
    const int wav  = tid >> 5;            // 0..15, each wave owns 32 hidden columns
    const int b0   = blockIdx.x * 16;     // batch tile

    const int hl      = lane >> 4;        // C/D layout: lanes 16-31 hold rows 8..15
    const int rowBase = 8 * hl;
    const int colIn   = lane & 15;
    const int col0    = wav * 32 + colIn; // hidden column of s=0 tile (s=1 adds 16)

    // ---- cell state in accumulator layout registers ----
    v8f cx[2];
    #pragma unroll
    for (int s = 0; s < 2; ++s)
        #pragma unroll
        for (int r = 0; r < 8; ++r)
            cx[s][r] = cx0[(size_t)(b0 + rowBase + r) * HID + col0 + 16 * s];

    // ---- initial h tile -> LDS f16 ----
    for (int idx = tid; idx < 16 * DIM; idx += 512) {
        const int row = idx >> 9, k = idx & (DIM - 1);
        hls[idx] = (_Float16)hx0[(size_t)(b0 + row) * HID + k];
    }
    __syncthreads();

    for (int t = 0; t < T_STEPS; ++t) {
        // Launder a zero *offset* (not the pointer!) through an empty asm:
        //  - LICM cannot hoist the (t-invariant) weight-fragment loads out of
        //    the time loop (no 8KB/lane scratch spill), but
        //  - the pointers keep their global address-space provenance, so the
        //    loads lower to global_load_b128 (LOADcnt only), not flat_load
        //    (which would tick DScnt and occupy the LDS path).
        size_t woff = 0;
        asm volatile("" : "+s"(woff));
        const _Float16* wih = wih_in + woff;
        const _Float16* whh = whh_in + woff;

        // keep = 1 - begin_mask[t]
        if (tid < 16) keeps[tid] = 1.0f - (float)mask[t * BATCH + b0 + tid];
        __syncthreads();   // also orders prev-step hls writes vs. this step's reads

        // h *= keep (in LDS), load+convert x_t -> LDS f16
        const float* xp = feat + ((size_t)t * BATCH + b0) * DIM;
        for (int idx = tid; idx < 16 * DIM; idx += 512) {
            const float kv = keeps[idx >> 9];
            hls[idx] = (_Float16)((float)hls[idx] * kv);
            xls[idx] = (_Float16)xp[idx];
        }
        // prefetch next timestep's x slab (128B per thread covers the 32KB tile)
        if (t + 1 < T_STEPS)
            __builtin_prefetch(feat + ((size_t)(t + 1) * BATCH + b0) * DIM + tid * 32, 0, 1);
        __syncthreads();

        // c *= keep (rows rowBase..rowBase+7 for this lane)
        #pragma unroll
        for (int r = 0; r < 8; ++r) {
            const float kv = keeps[rowBase + r];
            cx[0][r] *= kv;
            cx[1][r] *= kv;
        }

        // accumulators = bias (per column)
        v8f acc[4][2];
        #pragma unroll
        for (int g = 0; g < 4; ++g)
            #pragma unroll
            for (int s = 0; s < 2; ++s) {
                const float bv = bias[g * HID + col0 + 16 * s];
                #pragma unroll
                for (int r = 0; r < 8; ++r) acc[g][s][r] = bv;
            }

        // gates += x_t @ W_ih^T   (K = 512 over input dim)
        for (int kc = 0; kc < DIM; kc += 32) {
            const v16h a = load_a_frag(xls, kc, lane);
            #pragma unroll
            for (int g = 0; g < 4; ++g)
                #pragma unroll
                for (int s = 0; s < 2; ++s) {
                    const v16h b = load_b_frag(wih, g * HID + wav * 32 + s * 16, kc, lane);
                    acc[g][s] = __builtin_amdgcn_wmma_f32_16x16x32_f16(
                        false, a, false, b, (short)0, acc[g][s], false, false);
                }
        }
        // gates += h @ W_hh^T     (K = 512 over hidden dim)
        for (int kc = 0; kc < HID; kc += 32) {
            const v16h a = load_a_frag(hls, kc, lane);
            #pragma unroll
            for (int g = 0; g < 4; ++g)
                #pragma unroll
                for (int s = 0; s < 2; ++s) {
                    const v16h b = load_b_frag(whh, g * HID + wav * 32 + s * 16, kc, lane);
                    acc[g][s] = __builtin_amdgcn_wmma_f32_16x16x32_f16(
                        false, a, false, b, (short)0, acc[g][s], false, false);
                }
        }

        // nonlinearity + state update + outputs
        float hnew[2][8];
        #pragma unroll
        for (int s = 0; s < 2; ++s)
            #pragma unroll
            for (int r = 0; r < 8; ++r) {
                const float ig = sigmoidf_(acc[0][s][r]);
                const float fg = sigmoidf_(acc[1][s][r]);
                const float gg = tanhf(acc[2][s][r]);
                const float og = sigmoidf_(acc[3][s][r]);
                const float c  = fg * cx[s][r] + ig * gg;
                const float h  = og * tanhf(c);
                cx[s][r]   = c;
                hnew[s][r] = h;
                const size_t oi = ((size_t)t * BATCH + b0 + rowBase + r) * HID + col0 + 16 * s;
                hs_out[oi] = h;
                cs_out[oi] = c;
            }

        __syncthreads();   // all waves done reading hls before overwriting
        #pragma unroll
        for (int s = 0; s < 2; ++s)
            #pragma unroll
            for (int r = 0; r < 8; ++r)
                hls[(rowBase + r) * DIM + col0 + 16 * s] = (_Float16)hnew[s][r];
        // next iteration's first __syncthreads orders these writes vs. reads
    }
}

extern "C" void kernel_launch(void* const* d_in, const int* in_sizes, int n_in,
                              void* d_out, int out_size, void* d_ws, size_t ws_size,
                              hipStream_t stream) {
    (void)in_sizes; (void)n_in; (void)out_size; (void)ws_size;
    const float* feat = (const float*)d_in[0];
    const float* hx0  = (const float*)d_in[1];
    const float* cx0  = (const float*)d_in[2];
    const float* W_ih = (const float*)d_in[3];
    const float* W_hh = (const float*)d_in[4];
    const float* b_ih = (const float*)d_in[5];
    const float* b_hh = (const float*)d_in[6];
    const int*   bmsk = (const int*)d_in[7];

    // workspace layout: [W_ih f16 : 2MB][W_hh f16 : 2MB][bias f32 : 8KB]
    _Float16* wih16 = (_Float16*)d_ws;
    _Float16* whh16 = wih16 + (size_t)FH * DIM;
    float*    bias  = (float*)(whh16 + (size_t)FH * DIM);

    float* hs_out = (float*)d_out;
    float* cs_out = hs_out + (size_t)T_STEPS * BATCH * HID;

    lstm_prep_kernel<<<1024, 256, 0, stream>>>(W_ih, W_hh, b_ih, b_hh, wih16, whh16, bias);
    lstm_fused_kernel<<<BATCH / 16, 512, 0, stream>>>(
        feat, hx0, cx0, bmsk, wih16, whh16, bias, hs_out, cs_out);
}
